// BGATLayer_11862699672060
// MI455X (gfx1250) — matrix operand order repaired and checked by reading, not compiled, not run on hardware
//
#include <hip/hip_runtime.h>

#define ALPHA 0.2f
#define LOG2E 1.4426950408889634f

constexpr int Bx = 16;    // batch
constexpr int Nn = 1024;  // nodes
constexpr int Fd = 512;   // features (IN_F == OUT_F)

typedef __bf16 bf16;
typedef __attribute__((ext_vector_type(4)))  __bf16 v4bf;
typedef __attribute__((ext_vector_type(8)))  __bf16 v8bf;
typedef __attribute__((ext_vector_type(16))) __bf16 v16bf;
typedef __attribute__((ext_vector_type(8)))  float  v8f;
typedef __attribute__((ext_vector_type(4)))  float  v4f;

struct bfpair { v8bf lo, hi; };
static __device__ __forceinline__ v16bf joinbf(v8bf lo, v8bf hi) {
    bfpair p{lo, hi};
    return __builtin_bit_cast(v16bf, p);
}

// ---------------- pack x (f32 -> bf16), 4 elems/thread ----------------
__global__ __launch_bounds__(256) void pack_x_kernel(const float* __restrict__ x,
                                                     bf16* __restrict__ xb, int n4) {
    int i = blockIdx.x * blockDim.x + threadIdx.x;
    if (i < n4) {
        v4f v = reinterpret_cast<const v4f*>(x)[i];
        v4bf o;
        o[0] = (bf16)v[0]; o[1] = (bf16)v[1]; o[2] = (bf16)v[2]; o[3] = (bf16)v[3];
        reinterpret_cast<v4bf*>(xb)[i] = o;
    }
}

// ---------------- pack W transposed (f32 KxN -> bf16 NxK) ----------------
__global__ __launch_bounds__(256) void pack_wt_kernel(const float* __restrict__ W,
                                                      bf16* __restrict__ wt) {
    int idx = blockIdx.x * blockDim.x + threadIdx.x;   // over Fd*Fd
    int n = idx / Fd;
    int k = idx - n * Fd;
    wt[(size_t)n * Fd + k] = (bf16)W[(size_t)k * Fd + n];
}

// ---------------- GEMM1: h = x @ W  (bf16 WMMA, f32 acc) ----------------
// block = 256 threads (8 waves); tile 128 rows x 128 cols; wave w -> 16 rows, 8 col-tiles.
// Writes only transposed bf16 h (ht) and per-colblock s1/s2 partials (no f32 h at all).
__global__ __launch_bounds__(256) void gemm1_kernel(const bf16* __restrict__ xb,
                                                    const bf16* __restrict__ wt,
                                                    const float* __restrict__ a,
                                                    bf16* __restrict__ ht,
                                                    float* __restrict__ s1p,
                                                    float* __restrict__ s2p) {
    const int b      = blockIdx.z;
    const int rowblk = blockIdx.y * 128;
    const int colblk = blockIdx.x * 128;
    const int wave   = threadIdx.x >> 5;
    const int lane   = threadIdx.x & 31;
    const int lmod   = lane & 15;
    const int lhalf  = lane >> 4;

    const int row0 = rowblk + wave * 16;
    const bf16* arow = xb + ((size_t)b * Nn + row0 + lmod) * Fd;

    v8f acc[8];
#pragma unroll
    for (int t = 0; t < 8; ++t) acc[t] = (v8f)(0.0f);

    for (int k0 = 0; k0 < Fd; k0 += 32) {
        // A fragment: row = lmod, K = lhalf*16 + e
        v8bf alo = *reinterpret_cast<const v8bf*>(arow + k0 + lhalf * 16);
        v8bf ahi = *reinterpret_cast<const v8bf*>(arow + k0 + lhalf * 16 + 8);
        v16bf afrag = joinbf(alo, ahi);
#pragma unroll
        for (int t = 0; t < 8; ++t) {
            // B fragment: N = lmod, K = lhalf*16 + e  (contiguous in Wt row)
            const bf16* bcol = wt + (size_t)(colblk + t * 16 + lmod) * Fd + k0 + lhalf * 16;
            v16bf bfrag = joinbf(*reinterpret_cast<const v8bf*>(bcol),
                                 *reinterpret_cast<const v8bf*>(bcol + 8));
            acc[t] = __builtin_amdgcn_wmma_f32_16x16x32_bf16(
                false, afrag, false, bfrag, (short)0, acc[t], false, false);
        }
    }

    // ---- write transposed bf16 h: lane packs 8 consecutive rows at fixed col ----
#pragma unroll
    for (int t = 0; t < 8; ++t) {
        const int col = colblk + t * 16 + lmod;
        v8bf pk;
#pragma unroll
        for (int r = 0; r < 8; ++r) pk[r] = (bf16)acc[t][r];
        *reinterpret_cast<v8bf*>(ht + ((size_t)b * Fd + col) * Nn + row0 + lhalf * 8) = pk;
    }

    // ---- fused s1/s2 partials: per-row dot of this 128-col slice with a1/a2 ----
    float a1v[8], a2v[8];
#pragma unroll
    for (int t = 0; t < 8; ++t) {
        const int col = colblk + t * 16 + lmod;
        a1v[t] = a[col];
        a2v[t] = a[Fd + col];
    }
#pragma unroll
    for (int r = 0; r < 8; ++r) {
        float p1 = 0.0f, p2 = 0.0f;
#pragma unroll
        for (int t = 0; t < 8; ++t) {
            p1 = fmaf(acc[t][r], a1v[t], p1);
            p2 = fmaf(acc[t][r], a2v[t], p2);
        }
        // reduce across the 16 lanes (lmod) that share this row; stays within half-wave
#pragma unroll
        for (int off = 8; off; off >>= 1) {
            p1 += __shfl_xor(p1, off, 32);
            p2 += __shfl_xor(p2, off, 32);
        }
        if (lmod == 0) {
            const size_t slot = (size_t)blockIdx.x * (Bx * Nn) + (size_t)b * Nn
                              + row0 + lhalf * 8 + r;
            s1p[slot] = p1;   // one writer per slot -> deterministic
            s2p[slot] = p2;
        }
    }
}

// ---------------- combine the 4 per-colblock partials ----------------
__global__ __launch_bounds__(256) void combine_kernel(const float* __restrict__ s1p,
                                                      const float* __restrict__ s2p,
                                                      float* __restrict__ s1,
                                                      float* __restrict__ s2) {
    int i = blockIdx.x * blockDim.x + threadIdx.x;   // 0..Bx*Nn-1
    const int stride = Bx * Nn;
    float v1 = 0.0f, v2 = 0.0f;
#pragma unroll
    for (int p = 0; p < 4; ++p) {
        v1 += s1p[(size_t)p * stride + i];
        v2 += s2p[(size_t)p * stride + i];
    }
    s1[i] = v1;
    s2[i] = v2;
}

// ---------------- softmax stats: rowmax*log2e and 1/Z per (b,n) ----------------
__global__ __launch_bounds__(256) void m_kernel(const float* __restrict__ s1,
                                                const float* __restrict__ s2,
                                                float* __restrict__ mx2,
                                                float* __restrict__ invZ) {
    const int gwave = (blockIdx.x * blockDim.x + threadIdx.x) >> 5;  // 0..B*N-1
    const int lane  = threadIdx.x & 31;
    const int b     = gwave >> 10;
    const float s1v = s1[gwave];
    const float* s2b = s2 + (size_t)b * Nn;

    float mloc = -3.4e38f;
#pragma unroll 4
    for (int j = 0; j < 32; ++j) {
        float e = s1v + s2b[lane + 32 * j];
        e = e > 0.0f ? e : ALPHA * e;
        mloc = fmaxf(mloc, e);
    }
#pragma unroll
    for (int off = 16; off; off >>= 1) mloc = fmaxf(mloc, __shfl_xor(mloc, off, 32));
    const float m2 = mloc * LOG2E;

    float z = 0.0f;
#pragma unroll 4
    for (int j = 0; j < 32; ++j) {
        float e = s1v + s2b[lane + 32 * j];
        e = e > 0.0f ? e : ALPHA * e;
        z += __builtin_amdgcn_exp2f(fmaf(e, LOG2E, -m2));
    }
#pragma unroll
    for (int off = 16; off; off >>= 1) z += __shfl_xor(z, off, 32);

    if (lane == 0) { mx2[gwave] = m2; invZ[gwave] = 1.0f / z; }
}

// ---------------- GEMM2: out = elu( (softmax(e) @ h) + beta*h ) ----------------
// A fragments (attention probs) generated in-register; B and beta*h both from bf16 ht.
__global__ __launch_bounds__(256) void gemm2_kernel(const bf16* __restrict__ ht,
                                                    const float* __restrict__ s1,
                                                    const float* __restrict__ s2,
                                                    const float* __restrict__ mx2,
                                                    const float* __restrict__ invZ,
                                                    const float* __restrict__ beta_p,
                                                    float* __restrict__ out) {
    const int b      = blockIdx.z;
    const int rowblk = blockIdx.y * 128;
    const int colblk = blockIdx.x * 128;
    const int wave   = threadIdx.x >> 5;
    const int lane   = threadIdx.x & 31;
    const int lmod   = lane & 15;
    const int lhalf  = lane >> 4;

    const int arow = rowblk + wave * 16 + lmod;      // A-fragment row for this lane
    const float s1v = s1[(size_t)b * Nn + arow];
    const float m2  = mx2[(size_t)b * Nn + arow];
    const float* s2b = s2 + (size_t)b * Nn;

    v8f acc[8];
#pragma unroll
    for (int t = 0; t < 8; ++t) acc[t] = (v8f)(0.0f);

    for (int m0 = 0; m0 < Nn; m0 += 32) {
        // gather the 16 s2 values this lane's A elements need
        v4f sA0 = *reinterpret_cast<const v4f*>(s2b + m0 + lhalf * 8);
        v4f sA1 = *reinterpret_cast<const v4f*>(s2b + m0 + lhalf * 8 + 4);
        v4f sB0 = *reinterpret_cast<const v4f*>(s2b + m0 + 16 + lhalf * 8);
        v4f sB1 = *reinterpret_cast<const v4f*>(s2b + m0 + 16 + lhalf * 8 + 4);
        float sv[16];
#pragma unroll
        for (int i = 0; i < 4; ++i) {
            sv[i] = sA0[i]; sv[4 + i] = sA1[i]; sv[8 + i] = sB0[i]; sv[12 + i] = sB1[i];
        }
        v16bf afrag;
#pragma unroll
        for (int e = 0; e < 16; ++e) {
            float ev = s1v + sv[e];
            ev = ev > 0.0f ? ev : ALPHA * ev;
            afrag[e] = (bf16)__builtin_amdgcn_exp2f(fmaf(ev, LOG2E, -m2));
        }
#pragma unroll
        for (int t = 0; t < 8; ++t) {
            const bf16* bcol = ht + ((size_t)b * Fd + colblk + t * 16 + lmod) * Nn + m0 + lhalf * 8;
            v16bf bfrag = joinbf(*reinterpret_cast<const v8bf*>(bcol),
                                 *reinterpret_cast<const v8bf*>(bcol + 16));
            acc[t] = __builtin_amdgcn_wmma_f32_16x16x32_bf16(
                false, afrag, false, bfrag, (short)0, acc[t], false, false);
        }
    }

    const float beta = *beta_p;
    float izr[8];
#pragma unroll
    for (int r = 0; r < 8; ++r)
        izr[r] = invZ[(size_t)b * Nn + rowblk + wave * 16 + lhalf * 8 + r];

#pragma unroll
    for (int t = 0; t < 8; ++t) {
        const int col = colblk + t * 16 + lmod;
        // beta*h from transposed bf16 h: 8 consecutive rows at fixed col = one b128
        v8bf hvv = *reinterpret_cast<const v8bf*>(
            ht + ((size_t)b * Fd + col) * Nn + rowblk + wave * 16 + lhalf * 8);
#pragma unroll
        for (int r = 0; r < 8; ++r) {
            const int rowc = rowblk + wave * 16 + lhalf * 8 + r;
            float v = fmaf(acc[t][r], izr[r], beta * (float)hvv[r]);
            out[((size_t)b * Nn + rowc) * Fd + col] =
                v > 0.0f ? v : (__builtin_amdgcn_exp2f(v * LOG2E) - 1.0f);  // elu, alpha=1
        }
    }
}

extern "C" void kernel_launch(void* const* d_in, const int* in_sizes, int n_in,
                              void* d_out, int out_size, void* d_ws, size_t ws_size,
                              hipStream_t stream) {
    const float* x    = (const float*)d_in[0];  // (16,1024,512)
    const float* W    = (const float*)d_in[1];  // (512,512)
    const float* a    = (const float*)d_in[2];  // (1024,1)
    const float* beta = (const float*)d_in[3];  // (1,)

    char* ws = (char*)d_ws;
    const size_t sz_xb = (size_t)Bx * Nn * Fd * sizeof(bf16);  // 16 MB
    const size_t sz_wt = (size_t)Fd * Fd * sizeof(bf16);       // 0.5 MB
    const size_t sz_ht = sz_xb;                                // 16 MB
    const size_t nrow  = (size_t)Bx * Nn;

    bf16*  xb   = (bf16*)ws;
    bf16*  wt   = (bf16*)(ws + sz_xb);
    bf16*  ht   = (bf16*)(ws + sz_xb + sz_wt);
    float* s1p  = (float*)(ws + sz_xb + sz_wt + sz_ht);        // 4 * 64 KB
    float* s2p  = s1p + 4 * nrow;                              // 4 * 64 KB
    float* s1   = s2p + 4 * nrow;
    float* s2   = s1 + nrow;
    float* mx2  = s2 + nrow;
    float* invZ = mx2 + nrow;

    const int n4 = Bx * Nn * Fd / 4;
    pack_x_kernel<<<(n4 + 255) / 256, 256, 0, stream>>>(x, xb, n4);
    pack_wt_kernel<<<(Fd * Fd + 255) / 256, 256, 0, stream>>>(W, wt);

    dim3 gtile(Fd / 128, Nn / 128, Bx);  // (4, 8, 16) = 512 workgroups
    gemm1_kernel<<<gtile, 256, 0, stream>>>(xb, wt, a, ht, s1p, s2p);

    combine_kernel<<<(int)(nrow / 256), 256, 0, stream>>>(s1p, s2p, s1, s2);
    m_kernel<<<(int)(nrow / 8), 256, 0, stream>>>(s1, s2, mx2, invZ);

    gemm2_kernel<<<gtile, 256, 0, stream>>>(ht, s1, s2, mx2, invZ, beta, (float*)d_out);
}
